// DNC_84499186581787
// MI455X (gfx1250) — matrix operand order repaired and checked by reading, compiled, not run on hardware
//
#include <hip/hip_runtime.h>
#include <math.h>

// Problem sizes (match reference)
#define Bb   64
#define Nn   512
#define Vv   64
#define Hh   512
#define Rr   4
#define Aa   18
#define INn  128
#define TOPK 32
#define EPSf 1e-8f

typedef __attribute__((ext_vector_type(16))) _Float16 v16h;
typedef __attribute__((ext_vector_type(8)))  float    v8f;

__device__ __forceinline__ float sigf(float x) { return 1.f / (1.f + expf(-x)); }
__device__ __forceinline__ float softplusf(float x) {
    return fmaxf(x, 0.f) + log1pf(expf(-fabsf(x)));
}

// ---------------------------------------------------------------------------
// Generic WMMA linear: out[M=64, N] = X1[64,K1]·W1[N,K1]^T (+ X2[64,K2]·W2^T)
//                                     (+ bias1 + bias2)
// One wave (32 threads) per 16x16 output tile. K1/K2 multiples of 32.
// Fragment k order: contiguous, k = 16*hl + j in both A and B (wave32).
// ---------------------------------------------------------------------------
__global__ __launch_bounds__(32)
void k_wmma_linear(const float* __restrict__ X1, const float* __restrict__ W1, int K1,
                   const float* __restrict__ X2, const float* __restrict__ W2, int K2,
                   const float* __restrict__ bias1, const float* __restrict__ bias2,
                   float* __restrict__ out, int N)
{
    const int lane = threadIdx.x;
    const int hl   = lane >> 4;       // half-wave: 0 or 1
    const int lid  = lane & 15;
    const int m0   = blockIdx.y * 16;
    const int n0   = blockIdx.x * 16;
    const int m    = m0 + lid;        // A row loaded by this lane
    const int n    = n0 + lid;        // B column loaded by this lane
    const int nc   = (n < N) ? n : (N - 1);
    const float bmask = (n < N) ? 1.f : 0.f;

    v8f acc = {0.f, 0.f, 0.f, 0.f, 0.f, 0.f, 0.f, 0.f};

    for (int pair = 0; pair < 2; ++pair) {
        const float* X = pair ? X2 : X1;
        const float* W = pair ? W2 : W1;
        const int    K = pair ? K2 : K1;
        if (X == nullptr) continue;
        const float* xrow = X + (size_t)m  * K;
        const float* wrow = W + (size_t)nc * K;
        for (int k0 = 0; k0 < K; k0 += 32) {
            v16h a, b;
            // A 16x32 f16: lane -> row m, K = j + 16*hl
#pragma unroll
            for (int j = 0; j < 16; ++j) {
                const int ka = k0 + j + (hl ? 16 : 0);
                a[j] = (_Float16)xrow[ka];
            }
            // B 32x16 f16: lane -> column n, K = j + 16*hl
#pragma unroll
            for (int j = 0; j < 16; ++j) {
                const int kb = k0 + j + (hl ? 16 : 0);
                b[j] = (_Float16)(wrow[kb] * bmask);
            }
            acc = __builtin_amdgcn_wmma_f32_16x16x32_f16(
                      false, a, false, b, (short)0, acc, false, false);
        }
    }

    float bv = 0.f;
    if (n < N) {
        if (bias1) bv += bias1[n];
        if (bias2) bv += bias2[n];
    }
    // C/D layout: VGPR i -> m = m0 + i + 8*hl, column n = n0 + lid
#pragma unroll
    for (int i = 0; i < 8; ++i) {
        const int mm = m0 + i + (hl ? 8 : 0);
        if (n < N) out[(size_t)mm * N + n] = acc[i] + bv;
    }
}

// ---------------------------------------------------------------------------
// Fused fw/bw/w_r/psi via WMMA, one wave per (b, 16-column tile of n).
//   fw[b,r,n] = sum_k rw[b,r,k] * link[b,n,k]   (B_fw[k][n] = link[b,n,k])
//   bw[b,r,n] = sum_k rw[b,r,k] * link[b,k,n]   (B_bw[k][n] = link[b,k,n])
// A = rw[b] (R=4 rows, padded to 16 with zeros). B_bw loads are coalesced
// across lanes per k (16 consecutive n); B_fw loads vectorize per-lane.
// Epilogue: w_r = pi0*bw + pi1*cw + pi2*fw ; psi = prod_r (1 - w_r).
// r=0..3 live in acc VGPRs 0..3 of the hl==0 half-wave.
// ---------------------------------------------------------------------------
__global__ __launch_bounds__(32)
void k_fwbw_wmma(const float* __restrict__ link, const float* __restrict__ rw,
                 const float* __restrict__ rpi, const float* __restrict__ cw,
                 float* __restrict__ wr_out, float* __restrict__ psi)
{
    const int lane = threadIdx.x;
    const int hl   = lane >> 4;
    const int lid  = lane & 15;
    const int b    = blockIdx.y;
    const int n    = blockIdx.x * 16 + lid;

    const int   rr    = (lid < Rr) ? lid : (Rr - 1);
    const float amask = (lid < Rr) ? 1.f : 0.f;
    const float* arow  = rw   + ((size_t)b * Rr + rr) * Nn;   // rw[b,r,:]
    const float* lrow  = link + ((size_t)b * Nn + n) * Nn;    // link[b,n,:] (fw)
    const float* lbase = link + (size_t)b * Nn * Nn + n;      // link[b,k,n] (bw)

    v8f afw = {0.f, 0.f, 0.f, 0.f, 0.f, 0.f, 0.f, 0.f};
    v8f abw = {0.f, 0.f, 0.f, 0.f, 0.f, 0.f, 0.f, 0.f};

    for (int k0 = 0; k0 < Nn; k0 += 32) {
        v16h a, bf, bb;
#pragma unroll
        for (int j = 0; j < 16; ++j) {
            const int ka = k0 + j + (hl ? 16 : 0);
            a[j] = (_Float16)(arow[ka] * amask);
        }
#pragma unroll
        for (int j = 0; j < 16; ++j) {
            const int kb = k0 + j + (hl ? 16 : 0);
            bf[j] = (_Float16)lrow[kb];
            bb[j] = (_Float16)lbase[(size_t)kb * Nn];
        }
        afw = __builtin_amdgcn_wmma_f32_16x16x32_f16(
                  false, a, false, bf, (short)0, afw, false, false);
        abw = __builtin_amdgcn_wmma_f32_16x16x32_f16(
                  false, a, false, bb, (short)0, abw, false, false);
    }

    if (hl == 0) {
        float p = 1.f;
#pragma unroll
        for (int i = 0; i < Rr; ++i) {
            const int br = b * Rr + i;
            const float w = rpi[br * 3 + 0] * abw[i]
                          + rpi[br * 3 + 1] * cw[(size_t)br * Nn + n]
                          + rpi[br * 3 + 2] * afw[i];
            wr_out[(size_t)br * Nn + n] = w;
            p *= (1.f - w);
        }
        psi[b * Nn + n] = p;
    }
}

// ---------------------------------------------------------------------------
// rvec[b,r,v] = sum_k w_r[b,r,k] * memory[b,k,v] via WMMA.
// One wave per (b, 16-column tile of v). B[k][v] = mem[b,k,v] -> coalesced
// across lanes per k. Writes into xcat [x | rvec] at offset IN + r*V + v.
// ---------------------------------------------------------------------------
__global__ __launch_bounds__(32)
void k_rvec_wmma(const float* __restrict__ wr, const float* __restrict__ mem,
                 float* __restrict__ xcat)
{
    const int lane = threadIdx.x;
    const int hl   = lane >> 4;
    const int lid  = lane & 15;
    const int b    = blockIdx.y;
    const int v    = blockIdx.x * 16 + lid;

    const int   rr    = (lid < Rr) ? lid : (Rr - 1);
    const float amask = (lid < Rr) ? 1.f : 0.f;
    const float* arow  = wr  + ((size_t)b * Rr + rr) * Nn;
    const float* bbase = mem + (size_t)b * Nn * Vv + v;       // mem[b,k,v]

    v8f acc = {0.f, 0.f, 0.f, 0.f, 0.f, 0.f, 0.f, 0.f};

    for (int k0 = 0; k0 < Nn; k0 += 32) {
        v16h a, bmat;
#pragma unroll
        for (int j = 0; j < 16; ++j) {
            const int ka = k0 + j + (hl ? 16 : 0);
            a[j] = (_Float16)(arow[ka] * amask);
        }
#pragma unroll
        for (int j = 0; j < 16; ++j) {
            const int kb = k0 + j + (hl ? 16 : 0);
            bmat[j] = (_Float16)bbase[(size_t)kb * Vv];
        }
        acc = __builtin_amdgcn_wmma_f32_16x16x32_f16(
                  false, a, false, bmat, (short)0, acc, false, false);
    }

    if (hl == 0) {
#pragma unroll
        for (int i = 0; i < Rr; ++i)
            xcat[b * (INn + Rr * Vv) + INn + i * Vv + v] = acc[i];
    }
}

// ---------------------------------------------------------------------------
// Row norms of memory (B,N,V) -> (B,N)
// ---------------------------------------------------------------------------
__global__ void k_memnorm(const float* __restrict__ mem, float* __restrict__ mnorm)
{
    int t = blockIdx.x * blockDim.x + threadIdx.x;
    if (t >= Bb * Nn) return;
    const float* p = mem + (size_t)t * Vv;
    float s = 0.f;
    for (int v = 0; v < Vv; ++v) { float x = p[v]; s += x * x; }
    mnorm[t] = fmaxf(sqrtf(s), EPSf);
}

// ---------------------------------------------------------------------------
// Post-process read projection ro (B,R,68): key norm, softplus beta, softmax pi
// ---------------------------------------------------------------------------
__global__ void k_readpost(const float* __restrict__ ro, float* __restrict__ keynorm,
                           float* __restrict__ rbeta, float* __restrict__ rpi)
{
    int t = threadIdx.x;               // b*R + r, 256 threads
    const float* p = ro + t * 68;
    float s = 0.f;
    for (int v = 0; v < Vv; ++v) s += p[v] * p[v];
    keynorm[t] = fmaxf(sqrtf(s), EPSf);
    rbeta[t]   = softplusf(p[64]);
    float a = p[65], b = p[66], c = p[67];
    float mx = fmaxf(a, fmaxf(b, c));
    float ea = expf(a - mx), eb = expf(b - mx), ec = expf(c - mx);
    float inv = 1.f / (ea + eb + ec);
    rpi[t * 3 + 0] = ea * inv;
    rpi[t * 3 + 1] = eb * inv;
    rpi[t * 3 + 2] = ec * inv;
}

// ---------------------------------------------------------------------------
// Read content weights: cosine sim * beta, top-K(32) mask, softmax over N.
// One 512-thread block per (b,r). Unique ranks via stable tie-break.
// ---------------------------------------------------------------------------
__global__ __launch_bounds__(512)
void k_read_content(const float* __restrict__ mem, const float* __restrict__ ro,
                    const float* __restrict__ mnorm, const float* __restrict__ keynorm,
                    const float* __restrict__ rbeta, float* __restrict__ cw)
{
    __shared__ float s[Nn];
    __shared__ float kth, mx;
    int br = blockIdx.x; int b = br >> 2;
    int n  = threadIdx.x;
    const float* mrow = mem + ((size_t)b * Nn + n) * Vv;
    const float* krow = ro + br * 68;
    float d = 0.f;
    for (int v = 0; v < Vv; ++v) d += mrow[v] * krow[v];
    float sc = rbeta[br] * d / (mnorm[b * Nn + n] * keynorm[br]);
    s[n] = sc;
    __syncthreads();
    int rank = 0;
    for (int j = 0; j < Nn; ++j) {
        float o = s[j];
        rank += (o > sc) || (o == sc && j < n);
    }
    if (rank == TOPK - 1) kth = sc;
    if (rank == 0)        mx  = sc;
    __syncthreads();
    float e = (sc >= kth) ? expf(sc - mx) : 0.f;
    __syncthreads();
    s[n] = e;
    __syncthreads();
    for (int st = Nn / 2; st > 0; st >>= 1) {
        if (n < st) s[n] += s[n + st];
        __syncthreads();
    }
    cw[(size_t)br * Nn + n] = e / s[0];
}

// Copy x into the concatenated LSTM input buffer (layout [x | rvec])
__global__ void k_xcat(const float* __restrict__ x, float* __restrict__ xcat)
{
    int t = blockIdx.x * blockDim.x + threadIdx.x;
    if (t >= Bb * INn) return;
    int b = t / INn, j = t % INn;
    xcat[b * (INn + Rr * Vv) + j] = x[t];
}

// LSTM pointwise
__global__ void k_lstm(const float* __restrict__ gates, const float* __restrict__ c_in,
                       float* __restrict__ h_new, float* __restrict__ c_new)
{
    int t = blockIdx.x * blockDim.x + threadIdx.x;
    if (t >= Bb * Hh) return;
    int b = t / Hh, j = t % Hh;
    const float* g = gates + (size_t)b * 4 * Hh;
    float ig = sigf(g[j]);
    float fg = sigf(g[Hh + j]);
    float gg = tanhf(g[2 * Hh + j]);
    float og = sigf(g[3 * Hh + j]);
    float cn = fg * c_in[t] + ig * gg;
    c_new[t] = cn;
    h_new[t] = og * tanhf(cn);
}

// Post-process write projection wo (B,195)
__global__ __launch_bounds__(64)
void k_wpost(const float* __restrict__ wo, float* __restrict__ erase,
             float* __restrict__ addv, float* __restrict__ wkeynorm,
             float* __restrict__ wbeta, float* __restrict__ ga, float* __restrict__ gw)
{
    int b = blockIdx.x; int v = threadIdx.x;
    const float* p = wo + b * 195;
    erase[b * Vv + v] = sigf(p[Vv + v]);
    addv [b * Vv + v] = tanhf(p[2 * Vv + v]);
    if (v == 0) {
        float s = 0.f;
        for (int i = 0; i < Vv; ++i) s += p[i] * p[i];
        wkeynorm[b] = fmaxf(sqrtf(s), EPSf);
        wbeta[b]    = softplusf(p[192]);
        ga[b]       = sigf(p[193]);
        gw[b]       = sigf(p[194]);
    }
}

// Write content weights: full softmax over N (512-thread block per b)
__global__ __launch_bounds__(512)
void k_write_content(const float* __restrict__ mem, const float* __restrict__ wo,
                     const float* __restrict__ mnorm, const float* __restrict__ wkeynorm,
                     const float* __restrict__ wbeta, float* __restrict__ cww)
{
    __shared__ float s[Nn];
    int b = blockIdx.x; int n = threadIdx.x;
    const float* mrow = mem + ((size_t)b * Nn + n) * Vv;
    const float* key  = wo + b * 195;
    float d = 0.f;
    for (int v = 0; v < Vv; ++v) d += mrow[v] * key[v];
    float sc = wbeta[b] * d / (mnorm[b * Nn + n] * wkeynorm[b]);
    s[n] = sc; __syncthreads();
    for (int st = Nn / 2; st > 0; st >>= 1) {
        if (n < st) s[n] = fmaxf(s[n], s[n + st]);
        __syncthreads();
    }
    float mx = s[0]; __syncthreads();
    float e = expf(sc - mx);
    s[n] = e; __syncthreads();
    for (int st = Nn / 2; st > 0; st >>= 1) {
        if (n < st) s[n] += s[n + st];
        __syncthreads();
    }
    cww[b * Nn + n] = e / s[0];
}

// Allocation weighting: stable rank sort + cumprod in sorted order, unsorted out
__global__ __launch_bounds__(512)
void k_alloc(const float* __restrict__ usage, float* __restrict__ alloc)
{
    __shared__ float u[Nn];
    __shared__ float us[Nn];
    __shared__ float pp[Nn];
    int b = blockIdx.x; int n = threadIdx.x;
    float un = usage[b * Nn + n];
    u[n] = un; __syncthreads();
    int rank = 0;
    for (int j = 0; j < Nn; ++j) {
        float o = u[j];
        rank += (o < un) || (o == un && j < n);   // stable ascending argsort
    }
    us[rank] = un; __syncthreads();
    if (n == 0) {
        float run = 1.f;
        for (int i = 0; i < Nn; ++i) { pp[i] = run; run *= us[i]; }
    }
    __syncthreads();
    alloc[b * Nn + n] = (1.f - un) * pp[rank];
}

// write_w, usage_new, prec_new (needs block-wide sum of write_w)
__global__ __launch_bounds__(512)
void k_write_w(const float* __restrict__ alloc, const float* __restrict__ cww,
               const float* __restrict__ ga, const float* __restrict__ gw,
               const float* __restrict__ usage_in, const float* __restrict__ ww_in,
               const float* __restrict__ prec_in, const float* __restrict__ psi,
               float* __restrict__ write_w, float* __restrict__ usage_new,
               float* __restrict__ prec_new)
{
    __shared__ float s[Nn];
    int b = blockIdx.x; int n = threadIdx.x;
    float a = ga[b];
    float w = gw[b] * (a * alloc[b * Nn + n] + (1.f - a) * cww[b * Nn + n]);
    write_w[b * Nn + n] = w;
    s[n] = w; __syncthreads();
    for (int st = Nn / 2; st > 0; st >>= 1) {
        if (n < st) s[n] += s[n + st];
        __syncthreads();
    }
    float wsum = s[0];
    prec_new[b * Nn + n] = (1.f - wsum) * prec_in[b * Nn + n] + w;
    float ui = usage_in[b * Nn + n], wi = ww_in[b * Nn + n];
    usage_new[b * Nn + n] = (ui + wi - ui * wi) * psi[b * Nn + n];
}

// Memory erase/add update
__global__ void k_memnew(const float* __restrict__ mem, const float* __restrict__ ww,
                         const float* __restrict__ erase, const float* __restrict__ addv,
                         float* __restrict__ out)
{
    int t = blockIdx.x * blockDim.x + threadIdx.x;
    if (t >= Bb * Nn * Vv) return;
    int v = t & (Vv - 1);
    int n = (t >> 6) & (Nn - 1);
    int b = t >> 15;
    float w = ww[b * Nn + n];
    out[t] = mem[t] * (1.f - w * erase[b * Vv + v]) + w * addv[b * Vv + v];
}

// Link matrix update: float4 vectorized, diagonal zeroed (bandwidth hot spot)
__global__ void k_linknew(const float* __restrict__ link, const float* __restrict__ ww,
                          const float* __restrict__ prec_in, float* __restrict__ out)
{
    size_t t  = (size_t)blockIdx.x * blockDim.x + threadIdx.x;
    size_t i4 = t * 4;
    if (i4 >= (size_t)Bb * Nn * Nn) return;
    int b   = (int)(i4 >> 18);                 // N*N = 2^18
    int rem = (int)(i4 & ((Nn * Nn) - 1));
    int n   = rem >> 9;
    int m0  = rem & (Nn - 1);
    float wn = ww[b * Nn + n];
    const float4 l4 = *(const float4*)(link + i4);
    const float4 w4 = *(const float4*)(ww + b * Nn + m0);
    const float4 p4 = *(const float4*)(prec_in + b * Nn + m0);
    float4 o;
    o.x = (n == m0 + 0) ? 0.f : ((1.f - wn) * (1.f - w4.x) * l4.x + wn * p4.x);
    o.y = (n == m0 + 1) ? 0.f : ((1.f - wn) * (1.f - w4.y) * l4.y + wn * p4.y);
    o.z = (n == m0 + 2) ? 0.f : ((1.f - wn) * (1.f - w4.z) * l4.z + wn * p4.z);
    o.w = (n == m0 + 3) ? 0.f : ((1.f - wn) * (1.f - w4.w) * l4.w + wn * p4.w);
    *(float4*)(out + i4) = o;
}

// ---------------------------------------------------------------------------
extern "C" void kernel_launch(void* const* d_in, const int* in_sizes, int n_in,
                              void* d_out, int out_size, void* d_ws, size_t ws_size,
                              hipStream_t stream)
{
    (void)in_sizes; (void)n_in; (void)out_size; (void)ws_size;
    const float* x      = (const float*)d_in[0];
    const float* h      = (const float*)d_in[1];
    const float* c      = (const float*)d_in[2];
    const float* mem    = (const float*)d_in[3];
    const float* rw     = (const float*)d_in[4];
    const float* wwts   = (const float*)d_in[5];   // write_weights (old)
    const float* usage  = (const float*)d_in[6];
    const float* link   = (const float*)d_in[7];
    const float* prec   = (const float*)d_in[8];
    const float* W_ih   = (const float*)d_in[9];
    const float* W_hh   = (const float*)d_in[10];
    const float* b_ih   = (const float*)d_in[11];
    const float* b_hh   = (const float*)d_in[12];
    const float* W_read = (const float*)d_in[13];  // (R,V+4,H) flat = (272,512)
    const float* b_read = (const float*)d_in[14];
    const float* W_write= (const float*)d_in[15];  // (195,512)
    const float* b_write= (const float*)d_in[16];
    const float* W_pol  = (const float*)d_in[17];
    const float* b_pol  = (const float*)d_in[18];
    const float* W_val  = (const float*)d_in[19];
    const float* b_val  = (const float*)d_in[20];

    float* out = (float*)d_out;
    float* ws  = (float*)d_ws;

    // d_out layout (flat, return order)
    float* o_logits = out + 0;           // 64*18
    float* o_value  = out + 1152;        // 64
    float* o_h      = out + 1216;        // 64*512
    float* o_c      = out + 33984;       // 64*512
    float* o_mem    = out + 66752;       // 64*512*64
    float* o_wr     = out + 2163904;     // 64*4*512
    float* o_ww     = out + 2294976;     // 64*512
    float* o_usage  = out + 2327744;     // 64*512
    float* o_link   = out + 2360512;     // 64*512*512
    float* o_prec   = out + 19137728;    // 64*512

    // workspace layout (floats)
    float* w_ro    = ws + 0;        // 64*272
    float* w_mnorm = ws + 17408;    // 64*512
    float* w_knorm = ws + 50176;    // 256
    float* w_rbeta = ws + 50432;    // 256
    float* w_rpi   = ws + 50688;    // 768
    float* w_cw    = ws + 51456;    // 64*4*512
    float* w_psi   = ws + 182528;   // 64*512
    float* w_xcat  = ws + 215296;   // 64*384
    float* w_gates = ws + 239872;   // 64*2048
    float* w_wo    = ws + 370944;   // 64*195
    float* w_erase = ws + 383424;   // 64*64
    float* w_addv  = ws + 387520;   // 64*64
    float* w_wkn   = ws + 391616;   // 64
    float* w_wbeta = ws + 391680;   // 64
    float* w_ga    = ws + 391744;   // 64
    float* w_gw    = ws + 391808;   // 64
    float* w_cww   = ws + 391872;   // 64*512
    float* w_alloc = ws + 424640;   // 64*512 (end: 457408 floats ~1.75 MB)

    k_memnorm<<<(Bb * Nn + 255) / 256, 256, 0, stream>>>(mem, w_mnorm);

    // ro = h · W_read^T + b_read  (64 x 272, K=512)
    k_wmma_linear<<<dim3(17, 4), 32, 0, stream>>>(
        h, W_read, Hh, nullptr, nullptr, 0, b_read, nullptr, w_ro, Rr * (Vv + 4));
    k_readpost<<<1, 256, 0, stream>>>(w_ro, w_knorm, w_rbeta, w_rpi);

    k_read_content<<<Bb * Rr, Nn, 0, stream>>>(mem, w_ro, w_mnorm, w_knorm,
                                               w_rbeta, w_cw);

    // fw/bw GEMMs + w_r + psi, fused, WMMA (coalesced both link directions)
    k_fwbw_wmma<<<dim3(Nn / 16, Bb), 32, 0, stream>>>(link, rw, w_rpi, w_cw,
                                                      o_wr, w_psi);

    k_xcat<<<(Bb * INn + 255) / 256, 256, 0, stream>>>(x, w_xcat);

    // rvec = w_r · memory, WMMA
    k_rvec_wmma<<<dim3(Vv / 16, Bb), 32, 0, stream>>>(o_wr, mem, w_xcat);

    // gates = [x,rvec]·W_ih^T + h·W_hh^T + b_ih + b_hh  (64 x 2048, fused)
    k_wmma_linear<<<dim3(128, 4), 32, 0, stream>>>(
        w_xcat, W_ih, INn + Rr * Vv, h, W_hh, Hh, b_ih, b_hh, w_gates, 4 * Hh);
    k_lstm<<<(Bb * Hh + 255) / 256, 256, 0, stream>>>(w_gates, c, o_h, o_c);

    // wo = h_new · W_write^T + b_write  (64 x 195)
    k_wmma_linear<<<dim3(13, 4), 32, 0, stream>>>(
        o_h, W_write, Hh, nullptr, nullptr, 0, b_write, nullptr, w_wo, 195);
    k_wpost<<<Bb, 64, 0, stream>>>(w_wo, w_erase, w_addv, w_wkn, w_wbeta, w_ga, w_gw);

    k_write_content<<<Bb, Nn, 0, stream>>>(mem, w_wo, w_mnorm, w_wkn, w_wbeta, w_cww);
    k_alloc<<<Bb, Nn, 0, stream>>>(usage, w_alloc);
    k_write_w<<<Bb, Nn, 0, stream>>>(w_alloc, w_cww, w_ga, w_gw, usage, wwts, prec,
                                     w_psi, o_ww, o_usage, o_prec);

    k_memnew<<<(Bb * Nn * Vv + 255) / 256, 256, 0, stream>>>(mem, o_ww, w_erase,
                                                             w_addv, o_mem);
    k_linknew<<<(Bb * Nn * Nn / 4 + 255) / 256, 256, 0, stream>>>(link, o_ww, prec,
                                                                  o_link);

    // policy / value heads
    k_wmma_linear<<<dim3(2, 4), 32, 0, stream>>>(
        o_h, W_pol, Hh, nullptr, nullptr, 0, b_pol, nullptr, o_logits, Aa);
    k_wmma_linear<<<dim3(1, 4), 32, 0, stream>>>(
        o_h, W_val, Hh, nullptr, nullptr, 0, b_val, nullptr, o_value, 1);
}
